// RTCDet_1022202217379
// MI455X (gfx1250) — compile-verified, matrix operand order checked by reading, not run on hardware
//
#include <hip/hip_runtime.h>
#include <hip/hip_bf16.h>
#include <math.h>

typedef __attribute__((ext_vector_type(16))) _Float16 v16h;
typedef __attribute__((ext_vector_type(8)))  float    v8f;
typedef __attribute__((ext_vector_type(4)))  unsigned v4u;
typedef __attribute__((ext_vector_type(8)))  int      v8i_;
typedef __attribute__((ext_vector_type(4)))  int      v4i_;

#define NUM_CLASSES 80
#define TOPK_K      1000
#define CAP         4096
#define CONF_T      0.01f
#define NMS_T       0.5f
#define NPAD        3072      // 3 levels * 1000, padded to multiple of 16/32
#define NWORDS      96        // NPAD / 32
#define NHALF       192       // NPAD / 16
#define NBINS       1024
#define NREAL       3000

// ---------------- workspace layout (bytes) ----------------
#define OFF_HIST      0u           // 3 * 1024 u32
#define OFF_CNT       12288u       // 3 u32 (+pad)
#define OFF_THRESH    12304u       // 3 f32 (+pad)
#define OFF_CKEY      12320u       // 3 * 4096 f32
#define OFF_CIDX      61472u       // 3 * 4096 i32
#define OFF_CSCORE    110624u      // 3072 f32   (combined, concat order)
#define OFF_CLABEL    122912u      // 3072 i32
#define OFF_CBOX      135200u      // 3072*4 f32
#define OFF_SSCORE    184352u      // 3072 f32   (sorted order)
#define OFF_SLABEL    196640u      // 3072 i32
#define OFF_SBOX      208928u      // 3072*4 f32
#define OFF_SOBOX     258080u      // 3072*4 f32 (class-offset boxes)
#define OFF_SAREA     307232u      // 3072 f32
#define OFF_SUPP      319520u      // 3072*192 u16 suppression bit-matrix
#define OFF_KEEP      1499168u     // 96 u32

// ---------------- zero accumulators ----------------
__global__ void k_zero(unsigned* p, int n) {
  int i = blockIdx.x * blockDim.x + threadIdx.x;
  if (i < n) p[i] = 0u;
}

// ---------------- streaming logit histogram (bandwidth bound) ----------------
__global__ void k_hist(const float4* __restrict__ cls, int n4, unsigned* __restrict__ hist) {
  __shared__ unsigned h[NBINS];
  for (int i = threadIdx.x; i < NBINS; i += blockDim.x) h[i] = 0u;
  __syncthreads();
  int stride = gridDim.x * blockDim.x;
  for (int i = blockIdx.x * blockDim.x + threadIdx.x; i < n4; i += stride) {
    __builtin_prefetch(cls + i + 2 * stride, 0, 0);   // global_prefetch_b8
    float4 v = cls[i];
    #pragma unroll
    for (int q = 0; q < 4; q++) {
      float x = (q == 0) ? v.x : (q == 1) ? v.y : (q == 2) ? v.z : v.w;
      int b = (int)((x + 16.0f) * 32.0f);
      b = b < 0 ? 0 : (b > NBINS - 1 ? NBINS - 1 : b);
      atomicAdd(&h[b], 1u);
    }
  }
  __syncthreads();
  for (int i = threadIdx.x; i < NBINS; i += blockDim.x)
    if (h[i]) atomicAdd(&hist[i], h[i]);
}

// ---------------- suffix-scan histogram -> k-th largest logit bin ----------------
__global__ void k_thresh(const unsigned* __restrict__ hist, float* __restrict__ threshOut) {
  __shared__ unsigned s[NBINS];
  int t = threadIdx.x;
  s[t] = hist[t];
  __syncthreads();
  for (int off = 1; off < NBINS; off <<= 1) {
    unsigned v = (t + off < NBINS) ? s[t + off] : 0u;
    __syncthreads();
    s[t] += v;               // suffix sum: count of elements in bins >= t
    __syncthreads();
  }
  if ((s[t] >= (unsigned)TOPK_K) && (t == NBINS - 1 || s[t + 1] < (unsigned)TOPK_K))
    *threshOut = -16.0f + (float)t * (1.0f / 32.0f);
  if (t == 0 && s[0] < (unsigned)TOPK_K) *threshOut = -16.0f;
}

// ---------------- compact survivors (>= k-th bin lower edge) ----------------
__global__ void k_compact(const float* __restrict__ cls, int n, const float* __restrict__ threshp,
                          float* __restrict__ ck, int* __restrict__ ci, unsigned* __restrict__ counter) {
  float th = *threshp;
  int stride = gridDim.x * blockDim.x;
  for (int i = blockIdx.x * blockDim.x + threadIdx.x; i < n; i += stride) {
    float x = cls[i];
    if (x >= th) {
      unsigned pos = atomicAdd(counter, 1u);
      if (pos < (unsigned)CAP) { ck[pos] = x; ci[pos] = i; }
    }
  }
}

// ---------------- bitonic sort (desc score, asc idx tie-break == stable argsort(-s)) ------
__device__ __forceinline__ bool kv_before(float sa, int ia, float sb, int ib) {
  return (sa > sb) || (sa == sb && ia < ib);
}
__device__ void bitonic4096(float* key, int* idx) {
  const int n = 4096;
  for (int k = 2; k <= n; k <<= 1)
    for (int j = k >> 1; j > 0; j >>= 1) {
      __syncthreads();
      for (int i = threadIdx.x; i < n; i += blockDim.x) {
        int p = i ^ j;
        if (p > i) {
          bool up = ((i & k) == 0);
          float sa = key[i], sb = key[p];
          int   ia = idx[i], ib = idx[p];
          bool inOrder = kv_before(sa, ia, sb, ib);
          if (up ? !inOrder : inOrder) {
            key[i] = sb; key[p] = sa; idx[i] = ib; idx[p] = ia;
          }
        }
      }
    }
  __syncthreads();
}

// ---------------- per-level: sort candidates, sigmoid top-1000, decode + gather ----------
__global__ void k_level_emit(const float* __restrict__ candKey, const int* __restrict__ candIdx,
                             const unsigned* __restrict__ counter, const float* __restrict__ boxPred,
                             int levelOffset,
                             float* __restrict__ combScore, int* __restrict__ combLabel,
                             float* __restrict__ combBox) {
  __shared__ float sk[CAP];
  __shared__ int   si[CAP];
  unsigned cnt = *counter; if (cnt > (unsigned)CAP) cnt = CAP;
  for (int i = threadIdx.x; i < CAP; i += blockDim.x) {
    if (i < (int)cnt) { sk[i] = candKey[i]; si[i] = candIdx[i]; }
    else              { sk[i] = -3.4e38f;   si[i] = 0x7fffffff; }
  }
  __syncthreads();
  bitonic4096(sk, si);
  if (threadIdx.x < TOPK_K) {
    int s = threadIdx.x;
    bool have = (s < (int)cnt);
    float logit = sk[s];
    int   fi    = si[s];
    float score = have ? 1.0f / (1.0f + __expf(-logit)) : 0.0f;
    if (score <= CONF_T) score = 0.0f;            // valid-mask == zeroed score
    int label  = have ? (fi % NUM_CLASSES) : 0;
    int anchor = have ? (fi / NUM_CLASSES) : 0;
    int g = levelOffset + s;
    combScore[g] = score;
    combLabel[g] = label;
    #pragma unroll
    for (int q = 0; q < 4; q++)
      combBox[g * 4 + q] = have ? boxPred[anchor * 4 + q] : 0.0f;
  }
}

// ---------------- global sort of 3000 entries + offset-box / area prep -------------------
__global__ void k_global_sort(const float* __restrict__ combScore, const int* __restrict__ combLabel,
                              const float* __restrict__ combBox,
                              float* __restrict__ sScore, int* __restrict__ sLabel,
                              float* __restrict__ sBox, float* __restrict__ sOffBox,
                              float* __restrict__ sArea) {
  __shared__ float sk[CAP];
  __shared__ int   si[CAP];
  __shared__ float smax[1024];
  int tid = threadIdx.x;
  // jnp.max(bboxes) for the class-offset trick
  float m = 0.0f;
  for (int i = tid; i < NREAL * 4; i += 1024) m = fmaxf(m, combBox[i]);
  smax[tid] = m;
  __syncthreads();
  for (int off = 512; off > 0; off >>= 1) {
    if (tid < off) smax[tid] = fmaxf(smax[tid], smax[tid + off]);
    __syncthreads();
  }
  float maxv = smax[0];
  __syncthreads();
  for (int i = tid; i < CAP; i += 1024) {
    if (i < NREAL)     { sk[i] = combScore[i]; si[i] = i; }
    else if (i < NPAD) { sk[i] = 0.0f;         si[i] = i; }          // pad entries
    else               { sk[i] = -1.0f;        si[i] = 0x7fffffff; } // sort filler
  }
  __syncthreads();
  bitonic4096(sk, si);
  for (int i = tid; i < NPAD; i += 1024) {
    int src = si[i];
    float score, b0, b1, b2, b3; int label;
    if (src < NREAL) {
      score = sk[i]; label = combLabel[src];
      b0 = combBox[src * 4 + 0]; b1 = combBox[src * 4 + 1];
      b2 = combBox[src * 4 + 2]; b3 = combBox[src * 4 + 3];
    } else { score = 0.0f; label = -1; b0 = b1 = b2 = b3 = 0.0f; }
    sScore[i] = score; sLabel[i] = label;
    sBox[i * 4 + 0] = b0; sBox[i * 4 + 1] = b1; sBox[i * 4 + 2] = b2; sBox[i * 4 + 3] = b3;
    float ar = fmaxf(b2 - b0, 0.0f) * fmaxf(b3 - b1, 0.0f);   // offset-invariant area
    sArea[i] = ar;
    float off = (float)label * (maxv + 1.0f);
    sOffBox[i * 4 + 0] = b0 + off; sOffBox[i * 4 + 1] = b1 + off;
    sOffBox[i * 4 + 2] = b2 + off; sOffBox[i * 4 + 3] = b3 + off;
  }
}

// ---------------- suppression bit-matrix: TDM tile stage + WMMA union-base + ballot ------
// Column tile (128 boxes = 2KB, reused by 8 waves x 8 rows) is staged LDS-side by the
// Tensor Data Mover (1-D descriptor, data_size=4B, tensor_dim0=tile_dim0=512 dwords).
// union(i,j) = area_i + area_j - inter(i,j); the rank-2 term (area_i + area_j) for each
// 16x16 tile is one v_wmma_f32_16x16x32_f16:  A=[area_m,1,0..], B=[1,area_n,0..]^T.
// iou > T  <=>  inter*(1+T) > T*(area_i+area_j)   (division-free).
__global__ void k_supp(const float* __restrict__ offBox, const float* __restrict__ area,
                       unsigned short* __restrict__ supp) {
  __shared__ float rb[16][4], ra[16];
  __shared__ float cb[128][4], ca[128];
  int rowBase  = blockIdx.y * 16;
  int colBase0 = blockIdx.x * 128;
  int tid = threadIdx.x;
  if (tid < 32) {   // wave 0 issues the TDM load of the 128 column boxes into LDS
    unsigned ldsOff = (unsigned)(unsigned long long)(uintptr_t)&cb[0][0];
    unsigned long long ga = (unsigned long long)(uintptr_t)(offBox + colBase0 * 4);
    v4u g0;
    g0.x = 1u;                                            // count=1, user descriptor
    g0.y = ldsOff;                                        // lds_addr
    g0.z = (unsigned)(ga & 0xffffffffull);                // global_addr[31:0]
    g0.w = (unsigned)((ga >> 32) & 0x01ffffffull)         // global_addr[56:32]
           | (2u << 30);                                  // type=2 ("image")
    v8i_ g1;
    g1[0] = (int)(2u << 16);        // workgroup_mask=0, data_size=2 (4 bytes)
    g1[1] = (int)(512u << 16);      // tensor_dim0 = 512 (lo16)
    g1[2] = (int)(1u << 16);        // tensor_dim0 hi16=0, tensor_dim1 = 1 (lo16)
    g1[3] = (int)(512u << 16);      // tensor_dim1 hi16=0, tile_dim0 = 512
    g1[4] = 1;                      // tile_dim1 = 1, tile_dim2 = 0
    g1[5] = 512;                    // tensor_dim0_stride lo32 = 512
    g1[6] = (int)(512u << 16);      // stride0 hi16=0, tensor_dim1_stride lo16 = 512
    g1[7] = 0;
    v4i_ z4 = {};
#if __clang_major__ >= 23
    v8i_ z8 = {};
    __builtin_amdgcn_tensor_load_to_lds(g0, g1, z4, z4, z8, 0);
#else
    __builtin_amdgcn_tensor_load_to_lds(g0, g1, z4, z4, 0);
#endif
    __builtin_amdgcn_s_wait_tensorcnt(0);
  }
  if (tid < 16) {
    ra[tid] = area[rowBase + tid];
    #pragma unroll
    for (int q = 0; q < 4; q++) rb[tid][q] = offBox[(rowBase + tid) * 4 + q];
  }
  if (tid < 128) ca[tid] = area[colBase0 + tid];
  __syncthreads();
  int w = tid >> 5, lane = tid & 31;
  int colTile = blockIdx.x * 8 + w;
  int colBase = colTile * 16;
  if (colBase + 15 <= rowBase) {   // strictly lower-triangular tile: col>row impossible
    if (lane < 8) {
      supp[(rowBase + lane)     * NHALF + colTile] = 0;
      supp[(rowBase + lane + 8) * NHALF + colTile] = 0;
    }
    return;
  }
  // A 16x32 f16: lane<16 holds row M=lane, K=0,1 in VGPR0 halves; rest zero.
  v16h a = {}; v16h b = {};
  if (lane < 16) {
    a[0] = (_Float16)ra[lane];          a[1] = (_Float16)1.0f;
    b[0] = (_Float16)1.0f;              b[1] = (_Float16)ca[w * 16 + lane];
  }
  v8f c = {};
  c = __builtin_amdgcn_wmma_f32_16x16x32_f16(false, a, false, b, (short)0, c, false, false);
  // C layout: c[v] = areaSum(m = v + 8*(lane>=16), n = lane&15)
  int half = (lane >= 16) ? 1 : 0;
  int n = lane & 15;
  float cx1 = cb[w * 16 + n][0], cy1 = cb[w * 16 + n][1];
  float cx2 = cb[w * 16 + n][2], cy2 = cb[w * 16 + n][3];
  #pragma unroll
  for (int v = 0; v < 8; v++) {
    int m = v + half * 8;
    int row = rowBase + m, col = colBase + n;
    float ix1 = fmaxf(rb[m][0], cx1), iy1 = fmaxf(rb[m][1], cy1);
    float ix2 = fminf(rb[m][2], cx2), iy2 = fminf(rb[m][3], cy2);
    float inter = fmaxf(ix2 - ix1, 0.0f) * fmaxf(iy2 - iy1, 0.0f);
    bool pred = (col > row) && (inter * (1.0f + NMS_T) > NMS_T * c[v]);
    unsigned b32 = (unsigned)__ballot(pred);       // low16: row v, high16: row v+8
    if (lane == v) {                               // uniform ballot, one writer per row pair
      supp[(rowBase + v)     * NHALF + colTile] = (unsigned short)(b32 & 0xFFFFu);
      supp[(rowBase + v + 8) * NHALF + colTile] = (unsigned short)(b32 >> 16);
    }
  }
}

// ---------------- greedy NMS scan: one wave32, 3 keep-words per lane, zero barriers ------
__global__ void k_greedy(const float* __restrict__ sScore, const unsigned short* __restrict__ supp,
                         unsigned* __restrict__ keepOut) {
  int lane = threadIdx.x;                    // 32 threads
  unsigned kw[3];
  #pragma unroll
  for (int s = 0; s < 3; s++) {
    int wIdx = s * 32 + lane;
    unsigned bits = 0u;
    for (int bpos = 0; bpos < 32; bpos++)
      if (sScore[wIdx * 32 + bpos] > CONF_T) bits |= (1u << bpos);
    kw[s] = bits;                            // keep init = valid mask
  }
  const unsigned* supp32 = (const unsigned*)supp;   // rows are 384B-aligned
  for (int i = 0; i < NREAL; i++) {
    const unsigned* nextp = supp32 + (i + 1) * NWORDS;   // hide L2 latency of row i+1
    __builtin_prefetch(nextp + lane, 0, 0);
    __builtin_prefetch(nextp + 32 + lane, 0, 0);
    __builtin_prefetch(nextp + 64 + lane, 0, 0);
    int wi = i >> 5;
    unsigned theirs = (unsigned)__shfl((int)kw[wi >> 5], wi & 31, 32);
    if ((theirs >> (i & 31)) & 1u) {         // i survives -> suppress its row (cols > i only)
      const unsigned* rowp = supp32 + i * NWORDS;
      kw[0] &= ~rowp[lane];
      kw[1] &= ~rowp[32 + lane];
      kw[2] &= ~rowp[64 + lane];
    }
  }
  keepOut[lane] = kw[0]; keepOut[32 + lane] = kw[1]; keepOut[64 + lane] = kw[2];
}

// ---------------- final packed output: [boxes 3000x4 | scores 3000 | labels 3000] --------
__global__ void k_out(const float* __restrict__ sScore, const int* __restrict__ sLabel,
                      const float* __restrict__ sBox, const unsigned* __restrict__ keep,
                      float* __restrict__ out) {
  int i = blockIdx.x * blockDim.x + threadIdx.x;
  if (i >= NREAL) return;
  bool k = (keep[i >> 5] >> (i & 31)) & 1u;
  out[NREAL * 4 + i]     = k ? sScore[i] : 0.0f;
  out[NREAL * 5 + i]     = k ? (float)sLabel[i] : -1.0f;
  #pragma unroll
  for (int q = 0; q < 4; q++) out[i * 4 + q] = k ? sBox[i * 4 + q] : 0.0f;
}

extern "C" void kernel_launch(void* const* d_in, const int* in_sizes, int n_in,
                              void* d_out, int out_size, void* d_ws, size_t ws_size,
                              hipStream_t stream) {
  char* ws = (char*)d_ws;
  unsigned*       hist      = (unsigned*)(ws + OFF_HIST);
  unsigned*       counters  = (unsigned*)(ws + OFF_CNT);
  float*          thresh    = (float*)(ws + OFF_THRESH);
  float*          ckey      = (float*)(ws + OFF_CKEY);
  int*            cidx      = (int*)(ws + OFF_CIDX);
  float*          combScore = (float*)(ws + OFF_CSCORE);
  int*            combLabel = (int*)(ws + OFF_CLABEL);
  float*          combBox   = (float*)(ws + OFF_CBOX);
  float*          sScore    = (float*)(ws + OFF_SSCORE);
  int*            sLabel    = (int*)(ws + OFF_SLABEL);
  float*          sBox      = (float*)(ws + OFF_SBOX);
  float*          sOffBox   = (float*)(ws + OFF_SOBOX);
  float*          sArea     = (float*)(ws + OFF_SAREA);
  unsigned short* supp      = (unsigned short*)(ws + OFF_SUPP);
  unsigned*       keepW     = (unsigned*)(ws + OFF_KEEP);
  float*          out       = (float*)d_out;

  // re-zero accumulators every call (ws is not re-poisoned between replays)
  k_zero<<<(3 * NBINS + 3 + 255) / 256, 256, 0, stream>>>(hist, 3 * NBINS + 3);

  for (int l = 0; l < 3; l++) {
    const float* cls = (const float*)d_in[2 * l + 0];
    const float* box = (const float*)d_in[2 * l + 1];
    int MC = in_sizes[2 * l + 0];
    k_hist<<<2048, 256, 0, stream>>>((const float4*)cls, MC / 4, hist + l * NBINS);
    k_thresh<<<1, NBINS, 0, stream>>>(hist + l * NBINS, thresh + l);
    k_compact<<<1024, 256, 0, stream>>>(cls, MC, thresh + l,
                                        ckey + l * CAP, cidx + l * CAP, counters + l);
    k_level_emit<<<1, 1024, 0, stream>>>(ckey + l * CAP, cidx + l * CAP, counters + l,
                                         box, l * TOPK_K, combScore, combLabel, combBox);
  }

  k_global_sort<<<1, 1024, 0, stream>>>(combScore, combLabel, combBox,
                                        sScore, sLabel, sBox, sOffBox, sArea);
  k_supp<<<dim3(NPAD / 128, NPAD / 16), 256, 0, stream>>>(sOffBox, sArea, supp);
  k_greedy<<<1, 32, 0, stream>>>(sScore, supp, keepW);
  k_out<<<(NREAL + 255) / 256, 256, 0, stream>>>(sScore, sLabel, sBox, keepW, out);
}
